// QuantizedYOLOv3Tiny_76312978915930
// MI455X (gfx1250) — 8-bit matrix operand order repaired and checked by reading, compiled, not run on hardware
//
#include <hip/hip_runtime.h>


typedef int i32x8 __attribute__((ext_vector_type(8)));

__global__ __launch_bounds__(256) void wscale_kernel(const float* __restrict__ w,
                                                     int n, float* __restrict__ out) {
    __shared__ float red[256];
    float m = 0.0f;
    for (int i = threadIdx.x; i < n; i += 256) m = fmaxf(m, fabsf(w[i]));
    red[threadIdx.x] = m;
    __syncthreads();
    for (int s = 128; s > 0; s >>= 1) {
        if (threadIdx.x < s) red[threadIdx.x] = fmaxf(red[threadIdx.x], red[threadIdx.x + s]);
        __syncthreads();
    }
    if (threadIdx.x == 0) out[0] = fmaxf(red[0], 1e-8f);
}

__global__ __launch_bounds__(256) void wpack1_kernel(const float* __restrict__ w,
                                                     const float* __restrict__ scale,
                                                     signed char* __restrict__ out, int n) {
    int i = blockIdx.x * 256 + threadIdx.x;
    if (i < n) {
        float x = w[i] / scale[0];
        x = fminf(fmaxf(x, -1.0f), 1.0f);
        out[i] = (signed char)(int)rintf(x);
    }
}

__global__ __launch_bounds__(256) void wpack_kernel(const float* __restrict__ w,
                                                    const float* __restrict__ scale,
                                                    signed char* __restrict__ packA,
                                                    int OC, int IC, int khw,
                                                    int nChunks, int total) {
    float s = scale[0];
    int K = IC * khw;
    for (int idx = blockIdx.x * 256 + threadIdx.x; idx < total; idx += gridDim.x * 256) {
        int tc   = idx >> 10;
        int off  = idx & 1023;
        int lane = off >> 5;
        int v    = (off >> 2) & 7;
        int b    = off & 3;
        int tile = tc / nChunks;
        int c    = tc - tile * nChunks;
        int m    = lane & 15;
        int hi   = lane >> 4;
        int k = ((v >= 4) ? 32 : 0) + (hi << 4) + ((v & 3) << 2) + b
                + (c << 6);
        int oc = tile * 16 + m;
        signed char q = 0;
        if (oc < OC && k < K) {
            float x = w[(size_t)oc * K + k] / s;
            x = fminf(fmaxf(x, -1.0f), 1.0f);
            q = (signed char)(int)rintf(x);
        }
        packA[idx] = q;
    }
}

__global__ __launch_bounds__(256) void conv1_kernel(const float* __restrict__ x,
                                                    const signed char* __restrict__ wq,
                                                    const float* __restrict__ scale,
                                                    unsigned char* __restrict__ out,
                                                    int N, int H, int W) {
    size_t idx = (size_t)blockIdx.x * 256 + threadIdx.x;
    size_t total = (size_t)N * 8 * H * W;
    if (idx >= total) return;
    int px = (int)(idx % W);
    size_t t = idx / W;
    int py = (int)(t % H); t /= H;
    int oc = (int)(t % 8);
    int n  = (int)(t / 8);
    float acc = 0.0f;
    for (int ic = 0; ic < 3; ++ic)
        for (int dy = 0; dy < 3; ++dy) {
            int yy = py + dy - 1;
            if (yy < 0 || yy >= H) continue;
            for (int dx = 0; dx < 3; ++dx) {
                int xx = px + dx - 1;
                if (xx < 0 || xx >= W) continue;
                int q = wq[((oc * 3 + ic) * 3 + dy) * 3 + dx];
                if (q) acc += (float)q * x[(((size_t)n * 3 + ic) * H + yy) * W + xx];
            }
        }
    float v = acc * scale[0];
    v = fminf(fmaxf(v, 0.0f), 6.0f) * 0.5f;
    int a = (int)rintf(v);
    a = a < 0 ? 0 : (a > 3 ? 3 : a);
    out[idx] = (unsigned char)a;
}

__global__ __launch_bounds__(256) void pool_kernel(const unsigned char* __restrict__ in,
                                                   unsigned char* __restrict__ out,
                                                   int N, int C, int Ho, int Wo) {
    size_t idx = (size_t)blockIdx.x * 256 + threadIdx.x;
    size_t total = (size_t)N * C * Ho * Wo;
    if (idx >= total) return;
    int xo = (int)(idx % Wo);
    size_t t = idx / Wo;
    int yo = (int)(t % Ho); t /= Ho;
    int c  = (int)(t % C);
    int n  = (int)(t / C);
    int Hi = Ho * 2, Wi = Wo * 2;
    size_t base = (((size_t)n * C + c) * Hi + yo * 2) * Wi + xo * 2;
    unsigned char m0 = in[base],      m1 = in[base + 1];
    unsigned char m2 = in[base + Wi], m3 = in[base + Wi + 1];
    unsigned char m = m0 > m1 ? m0 : m1;
    if (m2 > m) m = m2;
    if (m3 > m) m = m3;
    out[idx] = m;
}

template <int IC, int KHW>
__device__ constexpr int lds_off(int k) {
    if (k >= IC * KHW) return -1;
    if (KHW == 9) {
        int ic = k / 9;
        int r  = k - ic * 9;
        return ic * 54 + (r / 3) * 18 + (r % 3);
    }
    return k * 16;
}

__device__ __forceinline__ void async_b8(const unsigned char* g, unsigned char* l) {
    unsigned lp = (unsigned)(unsigned long long)(void*)l;
    asm volatile("global_load_async_to_lds_b8 %0, %1, off" :: "v"(lp), "v"(g) : "memory");
}
__device__ __forceinline__ void wait_async0() {
    asm volatile("s_wait_asynccnt 0x0" ::: "memory");
}

template <int IC, int KHW, int NCHUNKS, int FINAL, int NTB>
__global__ __launch_bounds__(256) void conv_wmma_t(
    const unsigned char* __restrict__ act, const signed char* __restrict__ packA,
    const float* __restrict__ scale, unsigned char* __restrict__ out8,
    float* __restrict__ outf, int N, int OC, int H, int W, int xTiles) {

    constexpr int PATCH = (KHW == 9) ? IC * 54 : IC * 16;
    __shared__ unsigned char smem[8 * PATCH];

    const int lane = threadIdx.x & 31;
    const int wave = threadIdx.x >> 5;
    const int n15  = lane & 15;
    const int hi16 = lane >> 4;
    const int wid  = blockIdx.x * 8 + wave;
    if (wid >= N * H * xTiles) return;

    const int xT = wid % xTiles;
    const int t  = wid / xTiles;
    const int y  = t % H;
    const int n  = t / H;
    const int x0 = xT * 16;
    unsigned char* slice = smem + wave * PATCH;

    if (KHW == 9) {
        for (int idx = lane; idx < PATCH; idx += 32) {
            int ic = idx / 54;
            int r  = idx - ic * 54;
            int dy = r / 18;
            int dx = r - dy * 18;
            int yy = y + dy - 1, xx = x0 + dx - 1;
            if (yy >= 0 && yy < H && xx >= 0 && xx < W)
                async_b8(&act[(((size_t)n * IC + ic) * H + yy) * W + xx], &slice[idx]);
            else
                slice[idx] = 0;
        }
    } else {
        for (int idx = lane; idx < PATCH; idx += 32) {
            int ic = idx >> 4, dx = idx & 15, xx = x0 + dx;
            if (xx < W)
                async_b8(&act[(((size_t)n * IC + ic) * H + y) * W + xx], &slice[idx]);
            else
                slice[idx] = 0;
        }
    }
    wait_async0();

    const int nTiles = (OC + 15) >> 4;
    const int tile0  = blockIdx.y * NTB;
    i32x8 cfr[NTB];
#pragma unroll
    for (int tt = 0; tt < NTB; ++tt) {
        i32x8 z = {0, 0, 0, 0, 0, 0, 0, 0};
        cfr[tt] = z;
    }
    const signed char* aBase = packA + lane * 32;

#pragma unroll
    for (int c = 0; c < NCHUNKS; ++c) {
        i32x8 bfr;
#pragma unroll
        for (int v = 0; v < 8; ++v) {
            unsigned int word = 0;
#pragma unroll
            for (int b = 0; b < 4; ++b) {
                const int kc = ((v >> 2) << 5) + ((v & 3) << 2) + b + (c << 6);
                const int o0 = lds_off<IC, KHW>(kc);
                const int o1 = lds_off<IC, KHW>(kc + 16);
                unsigned int byte = 0;
                if (o0 >= 0 || o1 >= 0) {
                    int off = hi16 ? o1 : o0;
                    if (off >= 0) byte = slice[off + n15];
                }
                word |= byte << (8 * b);
            }
            bfr[v] = (int)word;
        }

#pragma unroll
        for (int tt = 0; tt < NTB; ++tt) {
            int tile = tile0 + tt;
            if (tile < nTiles) {
                const int4* ap = (const int4*)(aBase + ((size_t)tile * NCHUNKS + c) * 1024);
                int4 a0 = ap[0], a1 = ap[1];
                i32x8 afr;
                afr[0] = a0.x; afr[1] = a0.y; afr[2] = a0.z; afr[3] = a0.w;
                afr[4] = a1.x; afr[5] = a1.y; afr[6] = a1.z; afr[7] = a1.w;
                cfr[tt] = __builtin_amdgcn_wmma_i32_16x16x64_iu8(true, afr, false, bfr,
                                                                 cfr[tt], false, false);
            }
        }
    }

    const int px = x0 + n15;
    if (px >= W) return;
    const float s = scale[0];
#pragma unroll
    for (int tt = 0; tt < NTB; ++tt) {
        int tile = tile0 + tt;
        if (tile >= nTiles) continue;
#pragma unroll
        for (int r = 0; r < 8; ++r) {
            int oc = tile * 16 + r + hi16 * 8;
            if (oc >= OC) continue;
            float v = s * (float)cfr[tt][r];
            size_t o = (((size_t)n * OC + oc) * H + y) * W + px;
            if (FINAL) {
                float val = fminf(fmaxf(2.0f * v, 0.0f), 1.0f);
                outf[o] = rintf(val * 255.0f) * (1.0f / 255.0f);
            } else {
                int a = (int)rintf(v);
                a = a < 0 ? 0 : (a > 3 ? 3 : a);
                out8[o] = (unsigned char)a;
            }
        }
    }
}

extern "C" void kernel_launch(void* const* d_in, const int* in_sizes, int n_in,
                              void* d_out, int out_size, void* d_ws, size_t ws_size,
                              hipStream_t stream) {
    const float* x = (const float*)d_in[0];
    const float* W[10];
    for (int i = 0; i < 10; ++i) W[i] = (const float*)d_in[1 + i];

    char* ws = (char*)d_ws;
    float* scales    = (float*)ws;
    signed char* wq1 = (signed char*)(ws + 256);
    const size_t packOff[9] = {512, 2560, 4608, 10752, 31232, 88576, 288256, 304640, 361984};
    unsigned char* PP   = (unsigned char*)(ws + (1u << 20));
    unsigned char* actA = PP + 52428800;
    unsigned char* actB = actA + 13107200;

    for (int i = 0; i < 10; ++i)
        wscale_kernel<<<1, 256, 0, stream>>>(W[i], in_sizes[1 + i], scales + i);

    wpack1_kernel<<<1, 256, 0, stream>>>(W[0], scales, wq1, 216);

    struct PK { int OC, IC, khw, nChunks, nTiles; };
    const PK pk[9] = {
        {  8,   8, 9,  2,  1}, { 16,   8, 9,  2,  1}, { 32,  16, 9,  3,  2},
        { 56,  32, 9,  5,  4}, {104,  56, 9,  8,  7}, {208, 104, 9, 15, 13},
        { 56, 208, 1,  4,  4}, {104,  56, 9,  8,  7}, { 30, 104, 9, 15,  2}};
    for (int i = 0; i < 9; ++i) {
        int total = pk[i].nTiles * pk[i].nChunks * 1024;
        wpack_kernel<<<(total + 255) / 256, 256, 0, stream>>>(
            W[1 + i], scales + 1 + i, (signed char*)(ws + packOff[i]),
            pk[i].OC, pk[i].IC, pk[i].khw, pk[i].nChunks, total);
    }

    {
        size_t total = (size_t)16 * 8 * 640 * 640;
        conv1_kernel<<<(unsigned)((total + 255) / 256), 256, 0, stream>>>(
            x, wq1, scales, PP, 16, 640, 640);
        size_t pt = (size_t)16 * 8 * 320 * 320;
        pool_kernel<<<(unsigned)((pt + 255) / 256), 256, 0, stream>>>(PP, actA, 16, 8, 320, 320);
    }

    auto pool = [&](const unsigned char* in, unsigned char* o, int C, int Ho, int Wo) {
        size_t t = (size_t)16 * C * Ho * Wo;
        pool_kernel<<<(unsigned)((t + 255) / 256), 256, 0, stream>>>(in, o, 16, C, Ho, Wo);
    };

#define LAUNCH_CONV(ICv, KHWv, NCv, FINv, NTBv, inP, o8, of, si, pi, OCv, Hv, Wv)        \
    do {                                                                                 \
        int xT = ((Wv) + 15) / 16;                                                       \
        int tot = 16 * (Hv) * xT;                                                        \
        int nT = ((OCv) + 15) / 16;                                                      \
        dim3 g((tot + 7) / 8, (nT + (NTBv) - 1) / (NTBv));                               \
        conv_wmma_t<ICv, KHWv, NCv, FINv, NTBv><<<g, 256, 0, stream>>>(                  \
            inP, (const signed char*)(ws + packOff[pi]), scales + (si), o8, of,          \
            16, OCv, Hv, Wv, xT);                                                        \
    } while (0)

    LAUNCH_CONV(  8, 9,  2, 0, 1, actA, PP,   nullptr, 1, 0,   8, 320, 320); pool(PP, actB,  8, 160, 160);
    LAUNCH_CONV(  8, 9,  2, 0, 1, actB, PP,   nullptr, 2, 1,  16, 160, 160); pool(PP, actA, 16,  80,  80);
    LAUNCH_CONV( 16, 9,  3, 0, 2, actA, PP,   nullptr, 3, 2,  32,  80,  80); pool(PP, actB, 32,  40,  40);
    LAUNCH_CONV( 32, 9,  5, 0, 4, actB, PP,   nullptr, 4, 3,  56,  40,  40); pool(PP, actA, 56,  20,  20);
    LAUNCH_CONV( 56, 9,  8, 0, 7, actA, actB, nullptr, 5, 4, 104,  20,  20);
    LAUNCH_CONV(104, 9, 15, 0, 7, actB, actA, nullptr, 6, 5, 208,  20,  20);
    LAUNCH_CONV(208, 1,  4, 0, 4, actA, actB, nullptr, 7, 6,  56,  20,  20);
    LAUNCH_CONV( 56, 9,  8, 0, 7, actB, actA, nullptr, 8, 7, 104,  20,  20);
    LAUNCH_CONV(104, 9, 15, 1, 2, actA, nullptr, (float*)d_out, 9, 8, 30, 20, 20);
#undef LAUNCH_CONV
}
